// PlanNet_82806969467170
// MI455X (gfx1250) — compile-verified, matrix operand order checked by reading, not run on hardware
//
#include <hip/hip_runtime.h>

// ---------------------------------------------------------------------------
// Types / helpers
// ---------------------------------------------------------------------------
typedef __attribute__((ext_vector_type(16))) __bf16 v16bf;
typedef __attribute__((ext_vector_type(8)))  float  v8f;

#define NNODES 5000
#define NLINKS 20000
#define NPATHS 50000
#define MAXLEN 8

__device__ inline unsigned short bf_bits(float f) {
  union { float f; unsigned u; } x{f};
  unsigned r = x.u + 0x7fffu + ((x.u >> 16) & 1u);   // round-to-nearest-even
  return (unsigned short)(r >> 16);
}

// packed f32x2 -> bf16x2 (single v_cvt_pk_bf16_f32 when available)
__device__ inline unsigned pk_bf16(float a, float b) {
#if __has_builtin(__builtin_amdgcn_cvt_pk_bf16_f32)
  typedef __attribute__((ext_vector_type(2))) __bf16 v2bf;
  v2bf r = __builtin_amdgcn_cvt_pk_bf16_f32(a, b);
  return __builtin_bit_cast(unsigned, r);
#else
  return (unsigned)bf_bits(a) | ((unsigned)bf_bits(b) << 16);
#endif
}
__device__ inline unsigned short f2bf16(float a) {
#if __has_builtin(__builtin_amdgcn_cvt_pk_bf16_f32)
  return (unsigned short)(pk_bf16(a, a) & 0xffffu);
#else
  return bf_bits(a);
#endif
}

__device__ inline float fast_rcp(float x) {
#if __has_builtin(__builtin_amdgcn_rcpf)
  return __builtin_amdgcn_rcpf(x);
#else
  return 1.0f / x;
#endif
}
// sigmoid via v_exp + v_rcp (no IEEE div expansion)
__device__ inline float sigmoidf_(float x) {
  return fast_rcp(1.0f + __expf(-x));
}
// CDNA5 has a TANH transcendental; fall back to exp+rcp form
__device__ inline float tanh_fast(float x) {
#if __has_builtin(__builtin_amdgcn_tanhf)
  return __builtin_amdgcn_tanhf(x);
#else
  float e = __expf(2.0f * x);
  return 1.0f - 2.0f * fast_rcp(e + 1.0f);
#endif
}
__device__ inline float seluf_(float x) {
  const float sc = 1.0507009873554805f, al = 1.6732632423543772f;
  return sc * (x > 0.0f ? x : al * (__expf(x) - 1.0f));
}
__device__ inline void atomic_add_f32(float* p, float v) {
  __hip_atomic_fetch_add(p, v, __ATOMIC_RELAXED, __HIP_MEMORY_SCOPE_AGENT);
}

__device__ inline v8f wmma_bf16(v16bf a, v16bf b, v8f c) {
  // v_wmma_f32_16x16x32_bf16 : D = A(16x32) * B(32x16) + C(16x16 f32)
  return __builtin_amdgcn_wmma_f32_16x16x32_bf16(
      false, a, false, b, (short)0, c, false, false);
}

// A-fragment (16x32 bf16) from a bf16 LDS stage, row stride S (elements).
// Per lane: two contiguous 8-element (16B) runs -> 2 x ds_load_b128.
__device__ inline v16bf make_afrag_bf(const unsigned short* stage, int S,
                                      int kbase, int lane) {
  int lo = lane & 15, hi = lane >> 4;
  union { v16bf v; uint4 q[2]; } u;
  u.q[0] = *(const uint4*)&stage[lo * S + kbase + hi * 8];
  u.q[1] = *(const uint4*)&stage[lo * S + kbase + 16 + hi * 8];
  return u.v;
}

// B-fragments pre-packed in fragment order: tile*512 ushorts, lane*16.
__device__ inline v16bf load_bfrag(const unsigned short* pk, int tile, int lane) {
  return *((const v16bf*)(pk + (size_t)tile * 512 + lane * 16));
}

// ---------------------------------------------------------------------------
// Weight packing: f32 [K,N] row-major -> bf16 B-fragment tiles (kc, j)
// mode 0: src[k*N+n]     mode 1 (ECC refold, N==32): src[(k>>5)*1024+(k&31)*32+n]
// ---------------------------------------------------------------------------
__global__ void pack_b_kernel(const float* __restrict__ src,
                              unsigned short* __restrict__ dst,
                              int K, int N, int mode) {
  int ntile_n = N >> 4;
  int ntiles  = (K >> 5) * ntile_n;
  int idx = blockIdx.x * blockDim.x + threadIdx.x;
  if (idx >= ntiles * 32) return;
  int tile = idx >> 5, lane = idx & 31;
  int kc = tile / ntile_n, j = tile % ntile_n;
  int lo = lane & 15, hi = lane >> 4;
  unsigned short* out = dst + (size_t)tile * 512 + lane * 16;
#pragma unroll
  for (int e = 0; e < 16; ++e) {
    int v = e >> 1, half = e & 1;
    int k = kc * 32 + ((v >> 2) << 4) + (hi << 3) + ((v & 3) << 1) + half;
    int n = j * 16 + lo;
    float val = (mode == 0) ? src[k * N + n]
                            : src[(k >> 5) * 1024 + (k & 31) * 32 + n];
    out[e] = bf_bits(val);
  }
}

// ---------------------------------------------------------------------------
// State init / zero / f32->bf16 conversion
// ---------------------------------------------------------------------------
__global__ void zero_kernel(float* p, int n) {
  int i = blockIdx.x * blockDim.x + threadIdx.x;
  if (i < n) p[i] = 0.0f;
}
__global__ void init_state_kernel(const float* __restrict__ init, float* st,
                                  int n, int d) {
  int i = blockIdx.x * blockDim.x + threadIdx.x;
  if (i < n * d) st[i] = ((i % d) == 0) ? init[i / d] : 0.0f;
}
__global__ void init_path_kernel(const float* __restrict__ pi, float* ps, int n) {
  int i = blockIdx.x * blockDim.x + threadIdx.x;
  if (i < n * 64) {
    int p = i >> 6, c = i & 63;
    ps[i] = (c == 0) ? pi[p] : ((c == 1) ? pi[n + p] : 0.0f);
  }
}
__global__ void f32_to_bf16_kernel(const float* __restrict__ src,
                                   unsigned* __restrict__ dst, int npairs) {
  int i = blockIdx.x * blockDim.x + threadIdx.x;
  if (i < npairs) {
    float2 f = ((const float2*)src)[i];
    dst[i] = pk_bf16(f.x, f.y);
  }
}

// ---------------------------------------------------------------------------
// GRU path scan: one wave = 16 paths, 8 steps, 48 WMMA / step.
// Persistent per-wave state (h f32, biases) lives in LDS; the j-loop is NOT
// unrolled so live VGPRs stay < 256 (no vgpr-msb switching).
// ---------------------------------------------------------------------------
__global__ __launch_bounds__(32) void gru_kernel(
    const unsigned short* __restrict__ ls_bf, const unsigned short* __restrict__ ns_bf,
    float* __restrict__ path_state, float* __restrict__ m_out,
    const unsigned short* __restrict__ wxB, const unsigned short* __restrict__ whB,
    const float* __restrict__ gru_b,
    const int* __restrict__ links_to_paths, const int* __restrict__ nodes_to_paths) {
  __shared__ __align__(16) unsigned short xstage[16][64];
  __shared__ __align__(16) unsigned short hstage[16][64];
  __shared__ __align__(16) float hf[16][64];
  __shared__ float blds[4][64];
  __shared__ int lidx[16];

  int lane = threadIdx.x;
  int lo = lane & 15, hi = lane >> 4;
  int pbase = blockIdx.x * 16;

  // gate biases -> LDS (combined input+recurrent parts)
  for (int i = lane; i < 64; i += 32) {
    blds[0][i] = gru_b[i]        + gru_b[192 + i];   // z
    blds[1][i] = gru_b[64 + i]   + gru_b[256 + i];   // r
    blds[2][i] = gru_b[128 + i];                     // xh
    blds[3][i] = gru_b[320 + i];                     // hh
  }

  // load h rows: f32 stage + bf16 stage
  {
    const float2* ps2 = (const float2*)(path_state + (size_t)pbase * 64);
    float2* hf2 = (float2*)&hf[0][0];
    unsigned* hb = (unsigned*)&hstage[0][0];
#pragma unroll
    for (int i = 0; i < 16; ++i) {
      float2 f = ps2[lane + 32 * i];
      hf2[lane + 32 * i] = f;
      hb[lane + 32 * i] = pk_bf16(f.x, f.y);
    }
  }
  __syncthreads();

  for (int t = 0; t < MAXLEN; ++t) {
    // gather x_t = [link_row(32) | node_row(32)] per path, already bf16
    int e = (pbase + lo) * MAXLEN + t;
    if (hi == 0) {
      int li = links_to_paths[e];
      lidx[lo] = li;
      const uint4* s = (const uint4*)(ls_bf + (size_t)li * 32);
      uint4* d = (uint4*)&xstage[lo][0];
#pragma unroll
      for (int c = 0; c < 4; ++c) d[c] = s[c];
    } else {
      int ni = nodes_to_paths[e];
      const uint4* s = (const uint4*)(ns_bf + (size_t)ni * 32);
      uint4* d = (uint4*)&xstage[lo][32];
#pragma unroll
      for (int c = 0; c < 4; ++c) d[c] = s[c];
    }
    __syncthreads();

    v16bf xA0 = make_afrag_bf(&xstage[0][0], 64, 0, lane);
    v16bf xA1 = make_afrag_bf(&xstage[0][0], 64, 32, lane);
    v16bf hA0 = make_afrag_bf(&hstage[0][0], 64, 0, lane);
    v16bf hA1 = make_afrag_bf(&hstage[0][0], 64, 32, lane);
    __syncthreads();

#pragma unroll 1
    for (int j = 0; j < 4; ++j) {
      v8f accZ = {}, accR = {}, accX = {}, accH = {};
      // z-gate: cols [0,64): xz+hz fused in one accumulator chain
      accZ = wmma_bf16(xA0, load_bfrag(wxB, j, lane), accZ);
      accZ = wmma_bf16(xA1, load_bfrag(wxB, 12 + j, lane), accZ);
      accZ = wmma_bf16(hA0, load_bfrag(whB, j, lane), accZ);
      accZ = wmma_bf16(hA1, load_bfrag(whB, 12 + j, lane), accZ);
      // r-gate: cols [64,128)
      accR = wmma_bf16(xA0, load_bfrag(wxB, j + 4, lane), accR);
      accR = wmma_bf16(xA1, load_bfrag(wxB, 12 + j + 4, lane), accR);
      accR = wmma_bf16(hA0, load_bfrag(whB, j + 4, lane), accR);
      accR = wmma_bf16(hA1, load_bfrag(whB, 12 + j + 4, lane), accR);
      // candidate: cols [128,192), x and h parts kept separate (r gating)
      accX = wmma_bf16(xA0, load_bfrag(wxB, j + 8, lane), accX);
      accX = wmma_bf16(xA1, load_bfrag(wxB, 12 + j + 8, lane), accX);
      accH = wmma_bf16(hA0, load_bfrag(whB, j + 8, lane), accH);
      accH = wmma_bf16(hA1, load_bfrag(whB, 12 + j + 8, lane), accH);

      int col = j * 16 + lo;
      float bzv = blds[0][col], brv = blds[1][col];
      float bxv = blds[2][col], bhv = blds[3][col];
#pragma unroll
      for (int v = 0; v < 8; ++v) {
        int rr = v + 8 * hi;
        float hold = hf[rr][col];
        float z  = sigmoidf_(accZ[v] + bzv);
        float r  = sigmoidf_(accR[v] + brv);
        float c  = tanh_fast(accX[v] + bxv + r * (accH[v] + bhv));
        float hn = z * hold + (1.0f - z) * c;
        hf[rr][col] = hn;
        hstage[rr][col] = f2bf16(hn);
        int link = lidx[rr];
        atomic_add_f32(&m_out[(size_t)link * 64 + col], hn);
      }
    }
    __syncthreads();
  }

  // store final path_state from the f32 LDS stage
  {
    float2* dst = (float2*)(path_state + (size_t)pbase * 64);
    const float2* hf2 = (const float2*)&hf[0][0];
#pragma unroll
    for (int i = 0; i < 16; ++i) dst[lane + 32 * i] = hf2[lane + 32 * i];
  }
}

// ---------------------------------------------------------------------------
// Edge MLP: link_new = (concat[node(32)|link(32)|m(64)] @ W1 + b1) @ W2 + b2
// ---------------------------------------------------------------------------
__global__ __launch_bounds__(32) void edge_mlp_kernel(
    const unsigned short* __restrict__ ns_bf, const unsigned short* __restrict__ ls_bf,
    const float* __restrict__ m, const unsigned short* __restrict__ ew1B,
    const unsigned short* __restrict__ ew2B, const float* __restrict__ e_b1,
    const float* __restrict__ e_b2, const int* __restrict__ links_to_nodes,
    float* __restrict__ link_new) {
  __shared__ __align__(16) unsigned short cstage[16][128];
  __shared__ __align__(16) unsigned short h1stage[16][32];
  int lane = threadIdx.x, lo = lane & 15, hi = lane >> 4;
  int lbase = blockIdx.x * 16;
  int row = lbase + lo;

  if (hi == 0) {  // node + link halves, already bf16
    int nd = links_to_nodes[row];
    const uint4* sn = (const uint4*)(ns_bf + (size_t)nd * 32);
    const uint4* sl = (const uint4*)(ls_bf + (size_t)row * 32);
    uint4* d0 = (uint4*)&cstage[lo][0];
    uint4* d1 = (uint4*)&cstage[lo][32];
#pragma unroll
    for (int c = 0; c < 4; ++c) d0[c] = sn[c];
#pragma unroll
    for (int c = 0; c < 4; ++c) d1[c] = sl[c];
  } else {        // m half: f32 -> bf16 pairs
    const float2* sm = (const float2*)(m + (size_t)row * 64);
    unsigned* d = (unsigned*)&cstage[lo][64];
#pragma unroll
    for (int c = 0; c < 32; ++c) {
      float2 f = sm[c];
      d[c] = pk_bf16(f.x, f.y);
    }
  }
  __syncthreads();

  v8f acc0 = {}, acc1 = {};
#pragma unroll
  for (int kc = 0; kc < 4; ++kc) {
    v16bf A = make_afrag_bf(&cstage[0][0], 128, kc * 32, lane);
    acc0 = wmma_bf16(A, load_bfrag(ew1B, kc * 2 + 0, lane), acc0);
    acc1 = wmma_bf16(A, load_bfrag(ew1B, kc * 2 + 1, lane), acc1);
  }
#pragma unroll
  for (int j = 0; j < 2; ++j) {
    v8f a = j ? acc1 : acc0;
    float b = e_b1[j * 16 + lo];
#pragma unroll
    for (int v = 0; v < 8; ++v)
      h1stage[v + 8 * hi][j * 16 + lo] = f2bf16(a[v] + b);
  }
  __syncthreads();

  v16bf A1 = make_afrag_bf(&h1stage[0][0], 32, 0, lane);
  v8f o0 = {}, o1 = {};
  o0 = wmma_bf16(A1, load_bfrag(ew2B, 0, lane), o0);
  o1 = wmma_bf16(A1, load_bfrag(ew2B, 1, lane), o1);
#pragma unroll
  for (int j = 0; j < 2; ++j) {
    v8f o = j ? o1 : o0;
    float b = e_b2[j * 16 + lo];
#pragma unroll
    for (int v = 0; v < 8; ++v)
      link_new[(size_t)(lbase + v + 8 * hi) * 32 + j * 16 + lo] = o[v] + b;
  }
}

// ---------------------------------------------------------------------------
// ECC conv refolded: msg[e] = Z[e,:]@Wk' + n[e]@Bk, Z[e,c*32+i] = ls_c * n_i.
// ---------------------------------------------------------------------------
__global__ __launch_bounds__(32) void ecc_msg_kernel(
    const float* __restrict__ node_state, const float* __restrict__ link_new,
    const unsigned short* __restrict__ wkB, const unsigned short* __restrict__ bkB,
    const int* __restrict__ senders, const int* __restrict__ receivers,
    float* __restrict__ agg) {
  __shared__ __align__(16) float nst[16][32];
  __shared__ __align__(16) float lst[16][32];
  __shared__ int recv[16];
  int lane = threadIdx.x, lo = lane & 15, hi = lane >> 4;
  int lbase = blockIdx.x * 16;
  int row = lbase + lo;

  if (hi == 0) {
    int s = senders[row];
    recv[lo] = receivers[row];
    const float4* sp = (const float4*)(node_state + (size_t)s * 32);
    float4* d = (float4*)&nst[lo][0];
#pragma unroll
    for (int c = 0; c < 8; ++c) d[c] = sp[c];
  } else {
    const float4* sp = (const float4*)(link_new + (size_t)row * 32);
    float4* d = (float4*)&lst[lo][0];
#pragma unroll
    for (int c = 0; c < 8; ++c) d[c] = sp[c];
  }
  __syncthreads();

  // hoist this lane's 16 needed node values (fragment element order)
  float nr[16];
#pragma unroll
  for (int e = 0; e < 16; ++e) {
    int k = (e < 8) ? (hi * 8 + e) : (16 + hi * 8 + (e - 8));
    nr[e] = nst[lo][k];
  }

  v8f acc0 = {}, acc1 = {};
#pragma unroll 1
  for (int kc = 0; kc < 32; ++kc) {
    float s = lst[lo][kc];
    union { v16bf v; unsigned u[8]; } A;
#pragma unroll
    for (int v = 0; v < 8; ++v)
      A.u[v] = pk_bf16(s * nr[2 * v], s * nr[2 * v + 1]);
    acc0 = wmma_bf16(A.v, load_bfrag(wkB, kc * 2 + 0, lane), acc0);
    acc1 = wmma_bf16(A.v, load_bfrag(wkB, kc * 2 + 1, lane), acc1);
  }
  {  // bias contribution: n @ bk_mat
    union { v16bf v; unsigned u[8]; } A;
#pragma unroll
    for (int v = 0; v < 8; ++v) A.u[v] = pk_bf16(nr[2 * v], nr[2 * v + 1]);
    acc0 = wmma_bf16(A.v, load_bfrag(bkB, 0, lane), acc0);
    acc1 = wmma_bf16(A.v, load_bfrag(bkB, 1, lane), acc1);
  }
#pragma unroll
  for (int j = 0; j < 2; ++j) {
    v8f a = j ? acc1 : acc0;
#pragma unroll
    for (int v = 0; v < 8; ++v)
      atomic_add_f32(&agg[(size_t)recv[v + 8 * hi] * 32 + j * 16 + lo], a[v]);
  }
}

// ---------------------------------------------------------------------------
// Node update: node_new = agg + node_state @ Wroot + broot
// ---------------------------------------------------------------------------
__global__ __launch_bounds__(32) void node_update_kernel(
    const unsigned short* __restrict__ ns_bf, const float* __restrict__ agg,
    const unsigned short* __restrict__ wrootB, const float* __restrict__ broot,
    float* __restrict__ node_new, int n_nodes) {
  __shared__ __align__(16) unsigned short nst[16][32];
  int lane = threadIdx.x, lo = lane & 15, hi = lane >> 4;
  int nbase = blockIdx.x * 16;
  if (hi == 0) {
    int r = nbase + lo;
    if (r >= n_nodes) r = n_nodes - 1;
    const uint4* sp = (const uint4*)(ns_bf + (size_t)r * 32);
    uint4* d = (uint4*)&nst[lo][0];
#pragma unroll
    for (int c = 0; c < 4; ++c) d[c] = sp[c];
  }
  __syncthreads();
  v16bf A = make_afrag_bf(&nst[0][0], 32, 0, lane);
  v8f o0 = {}, o1 = {};
  o0 = wmma_bf16(A, load_bfrag(wrootB, 0, lane), o0);
  o1 = wmma_bf16(A, load_bfrag(wrootB, 1, lane), o1);
#pragma unroll
  for (int j = 0; j < 2; ++j) {
    v8f o = j ? o1 : o0;
    float b = broot[j * 16 + lo];
#pragma unroll
    for (int v = 0; v < 8; ++v) {
      int r = nbase + v + 8 * hi;
      if (r < n_nodes)
        node_new[(size_t)r * 32 + j * 16 + lo] =
            o[v] + b + agg[(size_t)r * 32 + j * 16 + lo];
    }
  }
}

// ---------------------------------------------------------------------------
// Readout: selu(ps@W1+b1) -> selu(@W2+b2) -> concat(r2, ps) @ f_w + f_b
// ---------------------------------------------------------------------------
__global__ __launch_bounds__(32) void readout_kernel(
    const float* __restrict__ path_state, const unsigned short* __restrict__ rw1B,
    const unsigned short* __restrict__ rw2B, const float* __restrict__ r_b1,
    const float* __restrict__ r_b2, const float* __restrict__ f_w,
    const float* __restrict__ f_b, float* __restrict__ out) {
  __shared__ __align__(16) float psf[16][64];          // f32 for final dot
  __shared__ __align__(16) unsigned short psb[16][64]; // bf16 for WMMA
  __shared__ __align__(16) unsigned short r1b[16][32];
  __shared__ __align__(16) float r2f[16][32];
  int lane = threadIdx.x, lo = lane & 15, hi = lane >> 4;
  int pbase = blockIdx.x * 16;

  const float2* s2 = (const float2*)(path_state + (size_t)pbase * 64);
  float2* df = (float2*)&psf[0][0];
  unsigned* db = (unsigned*)&psb[0][0];
#pragma unroll
  for (int i = 0; i < 16; ++i) {
    float2 f = s2[lane + 32 * i];
    df[lane + 32 * i] = f;
    db[lane + 32 * i] = pk_bf16(f.x, f.y);
  }
  __syncthreads();

  v16bf A0 = make_afrag_bf(&psb[0][0], 64, 0, lane);
  v16bf A1 = make_afrag_bf(&psb[0][0], 64, 32, lane);
  v8f a0 = {}, a1 = {};
  a0 = wmma_bf16(A0, load_bfrag(rw1B, 0, lane), a0);   // (kc0, j0)
  a0 = wmma_bf16(A1, load_bfrag(rw1B, 2, lane), a0);   // (kc1, j0)
  a1 = wmma_bf16(A0, load_bfrag(rw1B, 1, lane), a1);   // (kc0, j1)
  a1 = wmma_bf16(A1, load_bfrag(rw1B, 3, lane), a1);   // (kc1, j1)
#pragma unroll
  for (int j = 0; j < 2; ++j) {
    v8f a = j ? a1 : a0;
    float b = r_b1[j * 16 + lo];
#pragma unroll
    for (int v = 0; v < 8; ++v)
      r1b[v + 8 * hi][j * 16 + lo] = f2bf16(seluf_(a[v] + b));
  }
  __syncthreads();

  v16bf A2 = make_afrag_bf(&r1b[0][0], 32, 0, lane);
  v8f c0 = {}, c1 = {};
  c0 = wmma_bf16(A2, load_bfrag(rw2B, 0, lane), c0);
  c1 = wmma_bf16(A2, load_bfrag(rw2B, 1, lane), c1);
#pragma unroll
  for (int j = 0; j < 2; ++j) {
    v8f c = j ? c1 : c0;
    float b = r_b2[j * 16 + lo];
#pragma unroll
    for (int v = 0; v < 8; ++v)
      r2f[v + 8 * hi][j * 16 + lo] = seluf_(c[v] + b);
  }
  __syncthreads();

  if (hi == 0) {
    float acc = f_b[0];
    for (int c = 0; c < 32; ++c) acc += r2f[lo][c] * f_w[c];
    for (int c = 0; c < 64; ++c) acc += psf[lo][c] * f_w[32 + c];
    out[pbase + lo] = acc;
  }
}

// ---------------------------------------------------------------------------
// Host orchestration
// ---------------------------------------------------------------------------
extern "C" void kernel_launch(void* const* d_in, const int* in_sizes, int n_in,
                              void* d_out, int out_size, void* d_ws, size_t ws_size,
                              hipStream_t stream) {
  (void)in_sizes; (void)n_in; (void)out_size; (void)ws_size;
  const float* link_init = (const float*)d_in[0];
  const float* node_init = (const float*)d_in[1];
  const float* path_init = (const float*)d_in[2];
  const float* gru_wx    = (const float*)d_in[3];
  const float* gru_wh    = (const float*)d_in[4];
  const float* gru_b     = (const float*)d_in[5];
  const float* e_w1      = (const float*)d_in[6];
  const float* e_b1      = (const float*)d_in[7];
  const float* e_w2      = (const float*)d_in[8];
  const float* e_b2      = (const float*)d_in[9];
  const float* ecc_wk    = (const float*)d_in[10];
  const float* ecc_bk    = (const float*)d_in[11];
  const float* ecc_wroot = (const float*)d_in[12];
  const float* ecc_broot = (const float*)d_in[13];
  const float* r_w1      = (const float*)d_in[14];
  const float* r_b1      = (const float*)d_in[15];
  const float* r_w2      = (const float*)d_in[16];
  const float* r_b2      = (const float*)d_in[17];
  const float* f_w       = (const float*)d_in[18];
  const float* f_b       = (const float*)d_in[19];
  const int* links_to_paths = (const int*)d_in[22];
  const int* nodes_to_paths = (const int*)d_in[23];
  const int* links_to_nodes = (const int*)d_in[24];
  const int* senders        = (const int*)d_in[25];
  const int* receivers      = (const int*)d_in[26];

  // workspace layout (256B aligned chunks)
  char* w = (char*)d_ws;
  size_t off = 0;
  auto take = [&](size_t bytes) {
    char* p = w + off;
    off += (bytes + 255) & ~(size_t)255;
    return p;
  };
  float* ls_a = (float*)take((size_t)NLINKS * 32 * 4);
  float* ls_b = (float*)take((size_t)NLINKS * 32 * 4);
  float* ns_a = (float*)take((size_t)NNODES * 32 * 4);
  float* ns_b = (float*)take((size_t)NNODES * 32 * 4);
  float* ps   = (float*)take((size_t)NPATHS * 64 * 4);
  float* mbuf = (float*)take((size_t)NLINKS * 64 * 4);
  float* agg  = (float*)take((size_t)NNODES * 32 * 4);
  unsigned short* ls_bf = (unsigned short*)take((size_t)NLINKS * 32 * 2);
  unsigned short* ns_bf = (unsigned short*)take((size_t)NNODES * 32 * 2);
  unsigned short* wxB    = (unsigned short*)take(24 * 1024);
  unsigned short* whB    = (unsigned short*)take(24 * 1024);
  unsigned short* ew1B   = (unsigned short*)take(8 * 1024);
  unsigned short* ew2B   = (unsigned short*)take(2 * 1024);
  unsigned short* wkB    = (unsigned short*)take(64 * 1024);
  unsigned short* bkB    = (unsigned short*)take(2 * 1024);
  unsigned short* wrootB = (unsigned short*)take(2 * 1024);
  unsigned short* rw1B   = (unsigned short*)take(4 * 1024);
  unsigned short* rw2B   = (unsigned short*)take(2 * 1024);

  auto pg = [](int tiles) { return (tiles * 32 + 255) / 256; };
  pack_b_kernel<<<pg(24), 256, 0, stream>>>(gru_wx, wxB, 64, 192, 0);
  pack_b_kernel<<<pg(24), 256, 0, stream>>>(gru_wh, whB, 64, 192, 0);
  pack_b_kernel<<<pg(8), 256, 0, stream>>>(e_w1, ew1B, 128, 32, 0);
  pack_b_kernel<<<pg(2), 256, 0, stream>>>(e_w2, ew2B, 32, 32, 0);
  pack_b_kernel<<<pg(64), 256, 0, stream>>>(ecc_wk, wkB, 1024, 32, 1);
  pack_b_kernel<<<pg(2), 256, 0, stream>>>(ecc_bk, bkB, 32, 32, 0);
  pack_b_kernel<<<pg(2), 256, 0, stream>>>(ecc_wroot, wrootB, 32, 32, 0);
  pack_b_kernel<<<pg(4), 256, 0, stream>>>(r_w1, rw1B, 64, 32, 0);
  pack_b_kernel<<<pg(2), 256, 0, stream>>>(r_w2, rw2B, 32, 32, 0);

  init_state_kernel<<<(NLINKS * 32 + 255) / 256, 256, 0, stream>>>(link_init, ls_a, NLINKS, 32);
  init_state_kernel<<<(NNODES * 32 + 255) / 256, 256, 0, stream>>>(node_init, ns_a, NNODES, 32);
  init_path_kernel<<<(NPATHS * 64 + 255) / 256, 256, 0, stream>>>(path_init, ps, NPATHS);

  float* ls = ls_a; float* ls_alt = ls_b;
  float* ns = ns_a; float* ns_alt = ns_b;

  for (int it = 0; it < 2; ++it) {
    f32_to_bf16_kernel<<<(NLINKS * 16 + 255) / 256, 256, 0, stream>>>(
        ls, (unsigned*)ls_bf, NLINKS * 16);
    f32_to_bf16_kernel<<<(NNODES * 16 + 255) / 256, 256, 0, stream>>>(
        ns, (unsigned*)ns_bf, NNODES * 16);

    zero_kernel<<<(NLINKS * 64 + 255) / 256, 256, 0, stream>>>(mbuf, NLINKS * 64);
    gru_kernel<<<NPATHS / 16, 32, 0, stream>>>(ls_bf, ns_bf, ps, mbuf, wxB, whB,
                                               gru_b, links_to_paths, nodes_to_paths);
    edge_mlp_kernel<<<NLINKS / 16, 32, 0, stream>>>(ns_bf, ls_bf, mbuf, ew1B, ew2B,
                                                    e_b1, e_b2, links_to_nodes, ls_alt);
    zero_kernel<<<(NNODES * 32 + 255) / 256, 256, 0, stream>>>(agg, NNODES * 32);
    ecc_msg_kernel<<<NLINKS / 16, 32, 0, stream>>>(ns, ls_alt, wkB, bkB, senders,
                                                   receivers, agg);
    node_update_kernel<<<(NNODES + 15) / 16, 32, 0, stream>>>(ns_bf, agg, wrootB,
                                                              ecc_broot, ns_alt, NNODES);
    { float* t = ls; ls = ls_alt; ls_alt = t; }
    { float* t = ns; ns = ns_alt; ns_alt = t; }
  }

  readout_kernel<<<NPATHS / 16, 32, 0, stream>>>(ps, rw1B, rw2B, r_b1, r_b2, f_w,
                                                 f_b, (float*)d_out);
}
